// MultiHeadAttention_51908974739766
// MI455X (gfx1250) — matrix operand order repaired and checked by reading, not compiled, not run on hardware
//
#include <hip/hip_runtime.h>
#include <stdint.h>

// MI455X / gfx1250 multi-head attention forward.
// B=1, S=4096, D=1024, H=16, HD=64. bf16 WMMA (16x16x32, f32 accum),
// TDM (tensor_load_to_lds) double-buffered weight tiles in the projections.

#define S_LEN 4096
#define D_DIM 1024
#define NHEAD 16
#define HDIM  64

typedef __attribute__((ext_vector_type(16))) __bf16          v16bf;
typedef __attribute__((ext_vector_type(8)))  float           v8f;
typedef __attribute__((ext_vector_type(16))) unsigned short  v16u16;
typedef __attribute__((ext_vector_type(8)))  unsigned short  v8u16;
typedef __attribute__((ext_vector_type(4)))  unsigned int    v4u32;
typedef __attribute__((ext_vector_type(8)))  int             v8i32;
typedef __attribute__((ext_vector_type(4)))  int             v4i32;

union BF16x16 { v16u16 u; v16bf b; };

static __device__ __forceinline__ unsigned short f2bf(float f) {
  unsigned int u = __float_as_uint(f);
  u += 0x7FFFu + ((u >> 16) & 1u);          // round-to-nearest-even
  return (unsigned short)(u >> 16);
}

static __device__ __forceinline__ v8f wmma_bf16(v16bf a, v16bf b, v8f c) {
  return __builtin_amdgcn_wmma_f32_16x16x32_bf16(
      /*neg_a=*/false, a, /*neg_b=*/false, b,
      /*c_mod=*/(short)0, c, /*reuse_a=*/false, /*reuse_b=*/false);
}

// A-operand (16x32, MxK): lanes 0-15 row M=lane, halves = K {0..7,16..23};
// lanes 16-31 row M=lane-16, halves = K {8..15,24..31}.
static __device__ __forceinline__ v16bf load_a_frag(const unsigned short* src,
                                                    int row, int ld, int k0, int lane) {
  int koff = k0 + ((lane & 16) ? 8 : 0);
  const unsigned short* p = src + (size_t)row * ld + koff;
  v8u16 lo = *(const v8u16*)(p);
  v8u16 hi = *(const v8u16*)(p + 16);
  BF16x16 r;
#pragma unroll
  for (int i = 0; i < 8; ++i) { r.u[i] = lo[i]; r.u[i + 8] = hi[i]; }
  return r.b;
}

// B-operand (32x16, KxN), NT form (src = B^T row-major): lane n reads row n;
// lanes 0-15 K {0..7,16..23}, lanes 16-31 K {8..15,24..31}.
static __device__ __forceinline__ v16bf load_b_frag(const unsigned short* src,
                                                    int row, int ld, int k0, int lane) {
  int koff = k0 + ((lane & 16) ? 8 : 0);
  const unsigned short* p = src + (size_t)row * ld + koff;
  v8u16 lo = *(const v8u16*)(p);
  v8u16 hi = *(const v8u16*)(p + 16);
  BF16x16 r;
#pragma unroll
  for (int i = 0; i < 8; ++i) { r.u[i] = lo[i]; r.u[i + 8] = hi[i]; }
  return r.b;
}

// Same B-operand read, from an LDS tile of shape [64][128] bf16.
static __device__ __forceinline__ v16bf load_b_frag_lds(const unsigned short* tile,
                                                        int row, int k0l, int lane) {
  int koff = k0l + ((lane & 16) ? 8 : 0);
  const unsigned short* p = tile + row * 128 + koff;
  v8u16 lo = *(const v8u16*)(p);
  v8u16 hi = *(const v8u16*)(p + 16);
  BF16x16 r;
#pragma unroll
  for (int i = 0; i < 8; ++i) { r.u[i] = lo[i]; r.u[i + 8] = hi[i]; }
  return r.b;
}

// ---- Tensor Data Mover: 2D tile (tile_d0 x tile_d1 elements, 2B each) -> LDS.
// D# layout per CDNA5 ISA ch.8: group0 = {count, lds_addr, global_addr, type=2},
// group1 = {data_size, tensor dims, tile dims, dim0 stride}, groups 2/3 = 0 (2D).
// This toolchain exposes the 6-arg builtin: (v4u32, v8i32, v4i32, v4i32, v8i32, i32).
static __device__ __forceinline__ void tdm_load_2d_bf16(
    unsigned int lds_off_bytes, const unsigned short* gptr,
    unsigned int tensor_d0, unsigned int tensor_d1,
    unsigned int tile_d0, unsigned int tile_d1,
    unsigned int stride0_elems)
{
  unsigned long long ga = (unsigned long long)(uintptr_t)gptr;
  v4u32 g0;
  g0[0] = 1u;                                          // count=1 (user descriptor)
  g0[1] = lds_off_bytes;                               // lds_addr [63:32]
  g0[2] = (unsigned int)(ga & 0xFFFFFFFFu);            // global_addr[31:0]
  g0[3] = (unsigned int)((ga >> 32) & 0x01FFFFFFu)     // global_addr[56:32]
        | 0x80000000u;                                 // type=2 ("image")
  v8i32 g1;
  g1[0] = 0x10000;                                     // data_size=1 (2 bytes)
  g1[1] = (int)((tensor_d0 & 0xFFFFu) << 16);                                  // tensor_dim0[15:0]
  g1[2] = (int)(((tensor_d0 >> 16) & 0xFFFFu) | ((tensor_d1 & 0xFFFFu) << 16));// d0[31:16] | d1[15:0]
  g1[3] = (int)(((tensor_d1 >> 16) & 0xFFFFu) | ((tile_d0 & 0xFFFFu) << 16));  // d1[31:16] | tile_d0
  g1[4] = (int)(tile_d1 & 0xFFFFu);                    // tile_dim1 (tile_dim2=0)
  g1[5] = (int)stride0_elems;                          // tensor_dim0_stride[31:0]
  g1[6] = 0;
  g1[7] = 0;
  v4i32 z4 = {0, 0, 0, 0};
  v8i32 z8 = {0, 0, 0, 0, 0, 0, 0, 0};
  __builtin_amdgcn_tensor_load_to_lds(g0, g1, z4, z4, z8, 0);
}

static __device__ __forceinline__ float rmax16(float v) {
#pragma unroll
  for (int m = 1; m < 16; m <<= 1) v = fmaxf(v, __shfl_xor(v, m, 32));
  return v;
}
static __device__ __forceinline__ float rsum16(float v) {
#pragma unroll
  for (int m = 1; m < 16; m <<= 1) v += __shfl_xor(v, m, 32);
  return v;
}

// ---------------- prep: f32 -> bf16 convert / weight transpose ----------------

__global__ void prep_hidden_kernel(const float* __restrict__ hs,
                                   unsigned short* __restrict__ hb, int n) {
  int i = blockIdx.x * blockDim.x + threadIdx.x;
  if (i < n) hb[i] = f2bf(hs[i]);
}

// wt[n*D + k] = bf16(w[k*D + n])   (store W^T for NT GEMM)
__global__ void prep_wt_kernel(const float* __restrict__ w,
                               unsigned short* __restrict__ wt) {
  int i = blockIdx.x * blockDim.x + threadIdx.x;   // i = n*D + k
  int n = i >> 10, k = i & (D_DIM - 1);
  wt[i] = f2bf(w[k * D_DIM + n]);
}

// ---------------- QKV projection + RoPE epilogue ----------------
// grid (S/128, D/64, 3), block 256. 8 waves as 4(M)x2(N), each wave 32x32.
// Weight tile (64 x 128 bf16) TDM-staged into LDS, double buffered.
__global__ __launch_bounds__(256) void qkv_rope_kernel(
    const unsigned short* __restrict__ hid,    // [S][D] bf16
    const unsigned short* __restrict__ wq_t,   // [D][D] bf16 (transposed)
    const unsigned short* __restrict__ wk_t,
    const unsigned short* __restrict__ wv_t,
    const float* __restrict__ freqs,           // [S][32][2] (cos,sin)
    unsigned short* __restrict__ q_ws,         // [H][S][HD]
    unsigned short* __restrict__ k_ws,         // [H][S][HD]
    unsigned short* __restrict__ v_ws)         // [H][HD][S]  (transposed)
{
  __shared__ unsigned short btile[2][64][128];   // 2 x 16 KB weight tiles
  const int lane = threadIdx.x & 31;
  const int wave = threadIdx.x >> 5;
  const int z = blockIdx.z;
  const unsigned short* wt = (z == 0) ? wq_t : (z == 1) ? wk_t : wv_t;
  const int m0 = blockIdx.x * 128 + (wave >> 1) * 32;
  const int n0b = blockIdx.y * 64;               // block's N origin
  const int nw = (wave & 1) * 32;                // wave's N offset inside tile

  // Prologue: DMA first weight chunk.
  if (wave == 0) {
    tdm_load_2d_bf16((unsigned int)(size_t)&btile[0][0][0],
                     wt + (size_t)n0b * D_DIM + 0,
                     D_DIM, D_DIM, 128, 64, D_DIM);
    __builtin_amdgcn_s_wait_tensorcnt(0);
  }
  __syncthreads();

  v8f c[2][2] = {};
  int buf = 0;
  for (int kc = 0; kc < D_DIM; kc += 128) {
    if (wave == 0 && kc + 128 < D_DIM) {
      tdm_load_2d_bf16((unsigned int)(size_t)&btile[buf ^ 1][0][0],
                       wt + (size_t)n0b * D_DIM + kc + 128,
                       D_DIM, D_DIM, 128, 64, D_DIM);
    }
    const unsigned short* bt = &btile[buf][0][0];
#pragma unroll
    for (int kl = 0; kl < 128; kl += 32) {
      int k0 = kc + kl;
      v16bf a0 = load_a_frag(hid, m0 + (lane & 15), D_DIM, k0, lane);
      v16bf a1 = load_a_frag(hid, m0 + 16 + (lane & 15), D_DIM, k0, lane);
      v16bf b0 = load_b_frag_lds(bt, nw + (lane & 15), kl, lane);
      v16bf b1 = load_b_frag_lds(bt, nw + 16 + (lane & 15), kl, lane);
      c[0][0] = wmma_bf16(a0, b0, c[0][0]);
      c[0][1] = wmma_bf16(a0, b1, c[0][1]);
      c[1][0] = wmma_bf16(a1, b0, c[1][0]);
      c[1][1] = wmma_bf16(a1, b1, c[1][1]);
    }
    if (wave == 0) __builtin_amdgcn_s_wait_tensorcnt(0);
    __syncthreads();   // publishes next buffer; DS flush closes WAR window
    buf ^= 1;
  }

  // C layout: element r of lane L = row (r + 8*(L>=16)), col (L&15) of the tile.
#pragma unroll
  for (int mt = 0; mt < 2; ++mt) {
#pragma unroll
    for (int nt = 0; nt < 2; ++nt) {
#pragma unroll
      for (int r = 0; r < 8; ++r) {
        int row = m0 + mt * 16 + r + ((lane & 16) ? 8 : 0);   // seq position
        int col = n0b + nw + nt * 16 + (lane & 15);           // = h*64 + hd
        int h = col >> 6, hd = col & 63;
        float val = c[mt][nt][r];
        if (z == 2) {
          v_ws[(size_t)h * HDIM * S_LEN + (size_t)hd * S_LEN + row] = f2bf(val);
        } else {
          // RoPE: pair (hd even, hd odd) sits in adjacent lanes.
          float other = __shfl_xor(val, 1, 32);
          float2 cs = *(const float2*)(freqs + ((size_t)row * 32 + (hd >> 1)) * 2);
          float res = (hd & 1) ? fmaf(other,  cs.y, val * cs.x)    // a*sin + b*cos
                               : fmaf(other, -cs.y, val * cs.x);   // a*cos - b*sin
          unsigned short* dst = (z == 0) ? q_ws : k_ws;
          dst[(size_t)h * S_LEN * HDIM + (size_t)row * HDIM + hd] = f2bf(res);
        }
      }
    }
  }
}

// ---------------- flash attention (per head) ----------------
// grid (S/128, H), block 256. Each wave: 16 query rows, online softmax over
// key blocks of 32. Scores/probs staged through LDS to re-shape C->A layout.
__global__ __launch_bounds__(256) void flash_attn_kernel(
    const unsigned short* __restrict__ q_ws,   // [H][S][HD]
    const unsigned short* __restrict__ k_ws,   // [H][S][HD]
    const unsigned short* __restrict__ v_ws,   // [H][HD][S]
    const float* __restrict__ mask,            // [S]
    unsigned short* __restrict__ attn_ws)      // [S][D] bf16
{
  __shared__ unsigned short plds[8][16][48];   // per-wave 16x32 P tile (+pad)
  const int lane = threadIdx.x & 31;
  const int wave = threadIdx.x >> 5;
  const int h = blockIdx.y;
  const int q0 = blockIdx.x * 128 + wave * 16;
  const unsigned short* qh = q_ws + (size_t)h * S_LEN * HDIM;
  const unsigned short* kh = k_ws + (size_t)h * S_LEN * HDIM;
  const unsigned short* vh = v_ws + (size_t)h * HDIM * S_LEN;

  v16bf qf0 = load_a_frag(qh, q0 + (lane & 15), HDIM, 0, lane);
  v16bf qf1 = load_a_frag(qh, q0 + (lane & 15), HDIM, 32, lane);

  v8f o[4] = {};
  float rowm[8], rowl[8];
#pragma unroll
  for (int r = 0; r < 8; ++r) { rowm[r] = -1e30f; rowl[r] = 0.f; }
  const float scaling = 0.125f;   // HD^-0.5

  for (int kb = 0; kb < S_LEN; kb += 32) {
    v16bf kf00 = load_b_frag(kh, kb + (lane & 15), HDIM, 0, lane);
    v16bf kf01 = load_b_frag(kh, kb + (lane & 15), HDIM, 32, lane);
    v16bf kf10 = load_b_frag(kh, kb + 16 + (lane & 15), HDIM, 0, lane);
    v16bf kf11 = load_b_frag(kh, kb + 16 + (lane & 15), HDIM, 32, lane);
    v8f s0 = {}, s1 = {};
    s0 = wmma_bf16(qf0, kf00, s0);
    s0 = wmma_bf16(qf1, kf01, s0);
    s1 = wmma_bf16(qf0, kf10, s1);
    s1 = wmma_bf16(qf1, kf11, s1);

    float madd0 = (1.0f - mask[kb + (lane & 15)]) * -10000.0f;
    float madd1 = (1.0f - mask[kb + 16 + (lane & 15)]) * -10000.0f;

    float p0[8], p1[8];
#pragma unroll
    for (int r = 0; r < 8; ++r) {
      float a = s0[r] * scaling + madd0;
      float b = s1[r] * scaling + madd1;
      float bm = rmax16(fmaxf(a, b));             // per-row max (16-lane group)
      float mn = fmaxf(rowm[r], bm);
      float alpha = __expf(rowm[r] - mn);
      float e0 = __expf(a - mn);
      float e1 = __expf(b - mn);
      rowl[r] = rowl[r] * alpha + rsum16(e0 + e1);
      rowm[r] = mn;
      p0[r] = e0; p1[r] = e1;
#pragma unroll
      for (int t = 0; t < 4; ++t) o[t][r] *= alpha;
    }

    // Stage P (C layout) into LDS as bf16, reload as A operand.
    {
      int rbase = (lane & 16) ? 8 : 0;
#pragma unroll
      for (int r = 0; r < 8; ++r) {
        plds[wave][r + rbase][lane & 15]        = f2bf(p0[r]);
        plds[wave][r + rbase][16 + (lane & 15)] = f2bf(p1[r]);
      }
    }
    __syncthreads();   // uniform trip count across all waves
    v16bf pf;
    {
      const unsigned short* base = &plds[wave][lane & 15][0];
      int koff = (lane & 16) ? 8 : 0;
      v8u16 lo = *(const v8u16*)(base + koff);
      v8u16 hi = *(const v8u16*)(base + 16 + koff);
      BF16x16 rr;
#pragma unroll
      for (int i = 0; i < 8; ++i) { rr.u[i] = lo[i]; rr.u[i + 8] = hi[i]; }
      pf = rr.b;
    }
    __syncthreads();

    // O += P(16x32) * V_blk(32x64); v stored transposed -> NT loads.
#pragma unroll
    for (int t = 0; t < 4; ++t) {
      v16bf vf = load_b_frag(vh, t * 16 + (lane & 15), S_LEN, kb, lane);
      o[t] = wmma_bf16(pf, vf, o[t]);
    }
  }

#pragma unroll
  for (int r = 0; r < 8; ++r) {
    float inv = 1.0f / rowl[r];
    int row = q0 + r + ((lane & 16) ? 8 : 0);
#pragma unroll
    for (int t = 0; t < 4; ++t) {
      int d = h * HDIM + t * 16 + (lane & 15);
      attn_ws[(size_t)row * D_DIM + d] = f2bf(o[t][r] * inv);
    }
  }
}

// ---------------- output projection: out = attn @ wo (f32 out) ----------------
__global__ __launch_bounds__(256) void out_proj_kernel(
    const unsigned short* __restrict__ attn_ws,  // [S][D] bf16
    const unsigned short* __restrict__ wo_t,     // [D][D] bf16 transposed
    float* __restrict__ out)                     // [S][D] f32
{
  __shared__ unsigned short btile[2][64][128];
  const int lane = threadIdx.x & 31;
  const int wave = threadIdx.x >> 5;
  const int m0 = blockIdx.x * 128 + (wave >> 1) * 32;
  const int n0b = blockIdx.y * 64;
  const int nw = (wave & 1) * 32;

  if (wave == 0) {
    tdm_load_2d_bf16((unsigned int)(size_t)&btile[0][0][0],
                     wo_t + (size_t)n0b * D_DIM + 0,
                     D_DIM, D_DIM, 128, 64, D_DIM);
    __builtin_amdgcn_s_wait_tensorcnt(0);
  }
  __syncthreads();

  v8f c[2][2] = {};
  int buf = 0;
  for (int kc = 0; kc < D_DIM; kc += 128) {
    if (wave == 0 && kc + 128 < D_DIM) {
      tdm_load_2d_bf16((unsigned int)(size_t)&btile[buf ^ 1][0][0],
                       wo_t + (size_t)n0b * D_DIM + kc + 128,
                       D_DIM, D_DIM, 128, 64, D_DIM);
    }
    const unsigned short* bt = &btile[buf][0][0];
#pragma unroll
    for (int kl = 0; kl < 128; kl += 32) {
      int k0 = kc + kl;
      v16bf a0 = load_a_frag(attn_ws, m0 + (lane & 15), D_DIM, k0, lane);
      v16bf a1 = load_a_frag(attn_ws, m0 + 16 + (lane & 15), D_DIM, k0, lane);
      v16bf b0 = load_b_frag_lds(bt, nw + (lane & 15), kl, lane);
      v16bf b1 = load_b_frag_lds(bt, nw + 16 + (lane & 15), kl, lane);
      c[0][0] = wmma_bf16(a0, b0, c[0][0]);
      c[0][1] = wmma_bf16(a0, b1, c[0][1]);
      c[1][0] = wmma_bf16(a1, b0, c[1][0]);
      c[1][1] = wmma_bf16(a1, b1, c[1][1]);
    }
    if (wave == 0) __builtin_amdgcn_s_wait_tensorcnt(0);
    __syncthreads();
    buf ^= 1;
  }

#pragma unroll
  for (int mt = 0; mt < 2; ++mt)
#pragma unroll
    for (int nt = 0; nt < 2; ++nt)
#pragma unroll
      for (int r = 0; r < 8; ++r) {
        int row = m0 + mt * 16 + r + ((lane & 16) ? 8 : 0);
        int col = n0b + nw + nt * 16 + (lane & 15);
        out[(size_t)row * D_DIM + col] = c[mt][nt][r];
      }
}

// ---------------- host launcher ----------------
extern "C" void kernel_launch(void* const* d_in, const int* in_sizes, int n_in,
                              void* d_out, int out_size, void* d_ws, size_t ws_size,
                              hipStream_t stream) {
  const float* hidden = (const float*)d_in[0];
  const float* freqs  = (const float*)d_in[1];
  const float* mask   = (const float*)d_in[2];
  const float* wq = (const float*)d_in[3];
  const float* wk = (const float*)d_in[4];
  const float* wv = (const float*)d_in[5];
  const float* wo = (const float*)d_in[6];
  float* out = (float*)d_out;

  // Workspace layout (MiB offsets), total 48 MiB of bf16 staging.
  char* ws = (char*)d_ws;
  const size_t MB = 1u << 20;
  unsigned short* hid_bf = (unsigned short*)(ws + 0 * MB);    // 8 MiB [S][D]
  unsigned short* wq_t   = (unsigned short*)(ws + 8 * MB);    // 2 MiB
  unsigned short* wk_t   = (unsigned short*)(ws + 10 * MB);   // 2 MiB
  unsigned short* wv_t   = (unsigned short*)(ws + 12 * MB);   // 2 MiB
  unsigned short* wo_t   = (unsigned short*)(ws + 14 * MB);   // 2 MiB
  unsigned short* q_bf   = (unsigned short*)(ws + 16 * MB);   // 8 MiB [H][S][HD]
  unsigned short* k_bf   = (unsigned short*)(ws + 24 * MB);   // 8 MiB [H][S][HD]
  unsigned short* v_bf   = (unsigned short*)(ws + 32 * MB);   // 8 MiB [H][HD][S]
  unsigned short* a_bf   = (unsigned short*)(ws + 40 * MB);   // 8 MiB [S][D]

  const int nHid = S_LEN * D_DIM;
  prep_hidden_kernel<<<(nHid + 255) / 256, 256, 0, stream>>>(hidden, hid_bf, nHid);
  prep_wt_kernel<<<(D_DIM * D_DIM) / 256, 256, 0, stream>>>(wq, wq_t);
  prep_wt_kernel<<<(D_DIM * D_DIM) / 256, 256, 0, stream>>>(wk, wk_t);
  prep_wt_kernel<<<(D_DIM * D_DIM) / 256, 256, 0, stream>>>(wv, wv_t);
  prep_wt_kernel<<<(D_DIM * D_DIM) / 256, 256, 0, stream>>>(wo, wo_t);

  qkv_rope_kernel<<<dim3(S_LEN / 128, D_DIM / 64, 3), 256, 0, stream>>>(
      hid_bf, wq_t, wk_t, wv_t, freqs, q_bf, k_bf, v_bf);

  flash_attn_kernel<<<dim3(S_LEN / 128, NHEAD), 256, 0, stream>>>(
      q_bf, k_bf, v_bf, mask, a_bf);

  out_proj_kernel<<<dim3(S_LEN / 128, D_DIM / 64), 256, 0, stream>>>(
      a_bf, wo_t, out);
}
